// VertexConv_57415122813704
// MI455X (gfx1250) — matrix operand order repaired and checked by reading, not compiled, not run on hardware
//
#include <hip/hip_runtime.h>
#include <stdint.h>

#define NB 32
#define MM 128
#define EE 256
#define DD 128

typedef __attribute__((ext_vector_type(16))) __bf16 v16bf;
typedef __attribute__((ext_vector_type(8)))  float  v8f;

__device__ __forceinline__ unsigned short f2bf(float f) {
    union { float f; uint32_t u; } x; x.f = f;
    uint32_t r = x.u + 0x7FFFu + ((x.u >> 16) & 1u);   // RNE
    return (unsigned short)(r >> 16);
}

// ---------------------------------------------------------------------------
// S0[i,:] = softmax(bkk[i,:])  -> bf16.  128 blocks x 128 threads.
// ---------------------------------------------------------------------------
__global__ __launch_bounds__(128) void vc_s0(const float* __restrict__ bkk,
                                             unsigned short* __restrict__ S0) {
    __shared__ float red[128];
    const int i = blockIdx.x, j = threadIdx.x;
    float logit = bkk[i * MM + j];
    red[j] = logit; __syncthreads();
    for (int s = 64; s > 0; s >>= 1) { if (j < s) red[j] = fmaxf(red[j], red[j + s]); __syncthreads(); }
    float mx = red[0]; __syncthreads();
    float ex = __expf(logit - mx);
    red[j] = ex; __syncthreads();
    for (int s = 64; s > 0; s >>= 1) { if (j < s) red[j] += red[j + s]; __syncthreads(); }
    S0[i * MM + j] = f2bf(ex / red[0]);
}

// ---------------------------------------------------------------------------
// featT[n,d,j] = bf16(feature[n,j,d]).  Coalesced writes. 2048 x 256.
// ---------------------------------------------------------------------------
__global__ __launch_bounds__(256) void vc_feat_t(const float* __restrict__ feat,
                                                 unsigned short* __restrict__ fT) {
    int idx = blockIdx.x * 256 + threadIdx.x;        // < 32*128*128
    int n = idx >> 14;
    int r = idx & 16383;
    int d = r >> 7;
    int j = r & 127;
    fT[idx] = f2bf(feat[(n << 14) + (j << 7) + d]);
}

// ---------------------------------------------------------------------------
// S1[n,i,:] = softmax(feat[n,i,:]·Wkk[i,:,:]^T + bkk[i,:]) -> bf16.
// 4096 blocks (n*128+i) x 128 threads (j).
// ---------------------------------------------------------------------------
__global__ __launch_bounds__(128) void vc_s1(const float* __restrict__ feat,
                                             const float* __restrict__ Wkk,
                                             const float* __restrict__ bkk,
                                             unsigned short* __restrict__ S1) {
    __shared__ float frow[DD];
    __shared__ float red[128];
    const int blk = blockIdx.x;            // n*128 + i
    const int n = blk >> 7, i = blk & 127;
    const int j = threadIdx.x;

    frow[j] = feat[((size_t)n * MM + i) * DD + j];
    __syncthreads();

    const float4* wrow = reinterpret_cast<const float4*>(Wkk + ((size_t)i * MM + j) * DD);
    float dot = 0.f;
#pragma unroll 8
    for (int q = 0; q < DD / 4; ++q) {
        float4 wv = wrow[q];
        dot += frow[4 * q + 0] * wv.x + frow[4 * q + 1] * wv.y +
               frow[4 * q + 2] * wv.z + frow[4 * q + 3] * wv.w;
    }
    float logit = dot + bkk[i * MM + j];

    red[j] = logit; __syncthreads();
    for (int s = 64; s > 0; s >>= 1) { if (j < s) red[j] = fmaxf(red[j], red[j + s]); __syncthreads(); }
    float mx = red[0]; __syncthreads();
    float ex = __expf(logit - mx);
    red[j] = ex; __syncthreads();
    for (int s = 64; s > 0; s >>= 1) { if (j < s) red[j] += red[j + s]; __syncthreads(); }
    S1[(size_t)blk * MM + j] = f2bf(ex / red[0]);
}

// ---------------------------------------------------------------------------
// Main: per b=(n,e): mult[i,j] = (inc[n,i,e]?S1[n,i,j]:S0[i,j]) * inc[n,j,e]
//       t = mult @ feature[n]   (bf16 WMMA, fp32 acc)
//       out[n,e,i] = sum_d relu(t[i,d])*w1[d] + b1
// 8192 blocks x 256 threads (8 waves). LDS: 32KB mult + 32KB featT.
// ---------------------------------------------------------------------------
__global__ __launch_bounds__(256) void vc_main(const float* __restrict__ inc,
                                               const unsigned short* __restrict__ S1,
                                               const unsigned short* __restrict__ S0,
                                               const unsigned short* __restrict__ featT,
                                               const float* __restrict__ w1,
                                               const float* __restrict__ b1,
                                               float* __restrict__ out) {
    __shared__ __align__(16) unsigned short multL[MM * MM];   // 32 KB, (i, j)
    __shared__ __align__(16) unsigned short featL[DD * MM];   // 32 KB, featL[d*128+j]
    __shared__ __align__(16) unsigned short maskJ[MM];        // 0xFFFF / 0 per column
    __shared__ float incL[MM];

    const int b = blockIdx.x;
    const int n = b / EE;
    const int e = b % EE;
    const int t = threadIdx.x;

    if (t < MM) {
        float v = inc[(size_t)n * MM * EE + (size_t)t * EE + e];
        incL[t]  = v;
        maskJ[t] = (v != 0.0f) ? (unsigned short)0xFFFFu : (unsigned short)0;
    }
    __syncthreads();

    // featT slab for this n: b128 global -> b128 LDS (8 uint4 per thread)
    {
        const uint4* fsrc4 = reinterpret_cast<const uint4*>(featT + (size_t)n * DD * MM);
        uint4* featL4 = reinterpret_cast<uint4*>(featL);
#pragma unroll
        for (int k = t; k < (DD * MM) / 8; k += 256) featL4[k] = fsrc4[k];
    }

    // masked-softmax matrix, 8 bf16 per uint4 (8 per thread)
    {
        const uint4* s1n4 = reinterpret_cast<const uint4*>(S1 + (size_t)n * MM * MM);
        const uint4* s04  = reinterpret_cast<const uint4*>(S0);
        const uint4* mJ4  = reinterpret_cast<const uint4*>(maskJ);
        uint4* multL4 = reinterpret_cast<uint4*>(multL);
#pragma unroll
        for (int k = t; k < (MM * MM) / 8; k += 256) {
            int i = k >> 4;                       // 16 uint4 per 128-wide row
            uint4 v = (incL[i] != 0.0f) ? s1n4[k] : s04[k];
            uint4 m = mJ4[k & 15];
            v.x &= m.x; v.y &= m.y; v.z &= m.z; v.w &= m.w;
            multL4[k] = v;
        }
    }
    __syncthreads();

    const int lane = t & 31;
    const int w    = t >> 5;        // wave id: output rows [16w, 16w+16)
    const int mrow = lane & 15;     // A: M index / B,C: N index
    const int half = lane >> 4;     // 0/1

    float acc[8];
#pragma unroll
    for (int r = 0; r < 8; ++r) acc[r] = 0.f;

    for (int jt = 0; jt < 8; ++jt) {            // d-tile of output
        v8f c = {};
#pragma unroll
        for (int kt = 0; kt < 4; ++kt) {        // K=32 chunks over j
            // A fragment (mult): lane = (mrow, half);
            //   halfs 0..7  -> K = 32kt + 8*half + p
            //   halfs 8..15 -> K = 32kt + 16 + 8*half + (p-8)
            union { uint4 q[2]; v16bf v; } A;
            const uint4* ap = reinterpret_cast<const uint4*>(
                &multL[(16 * w + mrow) * MM + 32 * kt + 8 * half]);
            A.q[0] = ap[0];
            A.q[1] = ap[2];                      // +16 halfs
            // B fragment (featT): N = 16jt + mrow; K same as A
            union { uint4 q[2]; v16bf v; } B;
            const uint4* bp = reinterpret_cast<const uint4*>(
                &featL[(16 * jt + mrow) * MM + 32 * kt + 8 * half]);
            B.q[0] = bp[0];
            B.q[1] = bp[2];
            c = __builtin_amdgcn_wmma_f32_16x16x32_bf16(
                    false, A.v, false, B.v, (short)0, c, false, false);
        }
        // fused relu + w1-weighted pooling on the C fragment
        float w1v = w1[16 * jt + mrow];          // N = d column
#pragma unroll
        for (int r = 0; r < 8; ++r) {            // row i = 16w + 8*half + r
            float x = c[r];
            acc[r] += (x > 0.f ? x : 0.f) * w1v;
        }
    }

    // reduce over the 16 lanes sharing a row (xor masks stay within half-wave)
#pragma unroll
    for (int r = 0; r < 8; ++r) {
        float v = acc[r];
        v += __shfl_xor(v, 1, 32);
        v += __shfl_xor(v, 2, 32);
        v += __shfl_xor(v, 4, 32);
        v += __shfl_xor(v, 8, 32);
        acc[r] = v;
    }

    if (mrow == 0) {
        float bb = b1[0];
        float* dst = out + ((size_t)n * EE + e) * MM;
#pragma unroll
        for (int r = 0; r < 8; ++r)
            dst[16 * w + 8 * half + r] = acc[r] + bb;
    }
}

// ---------------------------------------------------------------------------
extern "C" void kernel_launch(void* const* d_in, const int* in_sizes, int n_in,
                              void* d_out, int out_size, void* d_ws, size_t ws_size,
                              hipStream_t stream) {
    const float* feat = (const float*)d_in[0];   // (32,128,128)
    const float* inc  = (const float*)d_in[1];   // (32,128,256)
    const float* Wkk  = (const float*)d_in[2];   // (128,128,128)
    const float* bkk  = (const float*)d_in[3];   // (128,128)
    const float* w1   = (const float*)d_in[4];   // (128)
    const float* b1   = (const float*)d_in[5];   // scalar
    float* out = (float*)d_out;                  // (32,256,128)

    char* ws = (char*)d_ws;
    unsigned short* S1 = (unsigned short*)ws;                                  // 32*128*128 bf16 (1 MB)
    unsigned short* S0 = (unsigned short*)(ws + (size_t)NB * MM * MM * 2);     // 128*128 bf16
    unsigned short* fT = (unsigned short*)(ws + (size_t)NB * MM * MM * 2 + (size_t)MM * MM * 2);

    vc_s0    <<<MM,              128, 0, stream>>>(bkk, S0);
    vc_feat_t<<<(NB*MM*DD)/256,  256, 0, stream>>>(feat, fT);
    vc_s1    <<<NB * MM,         128, 0, stream>>>(feat, Wkk, bkk, S1);
    vc_main  <<<NB * EE,         256, 0, stream>>>(inc, S1, S0, fT, w1, b1, out);
}
